// BertSelfAttention_45028437131524
// MI455X (gfx1250) — matrix operand order repaired and checked by reading, not compiled, not run on hardware
//
#include <hip/hip_runtime.h>

// ---------------------------------------------------------------------------
// BERT self-attention for MI455X (gfx1250, wave32, WMMA bf16 path)
//   B=4, S=2048, H=768, NH=12, HS=64
// Pipeline: f32->bf16 convert -> QKV GEMMs (head-major out) ->
//           flash-style streaming attention -> output projection GEMM.
// All matmuls use v_wmma_f32_16x16x32_bf16 (fp32 accumulate).
// B-operands are staged TRANSPOSED in LDS so every fragment read is two
// contiguous ds_load_b128; A/K tiles use gfx1250 async global->LDS copies.
// ---------------------------------------------------------------------------

typedef __attribute__((ext_vector_type(16))) __bf16 v16bf;
typedef __attribute__((ext_vector_type(8)))  float  v8f;

#if defined(__gfx1250__) &&                                             \
    __has_builtin(__builtin_amdgcn_global_load_async_to_lds_b128) &&    \
    __has_builtin(__builtin_amdgcn_s_wait_asynccnt)
#define USE_ASYNC_LDS 1
#else
#define USE_ASYNC_LDS 0
#endif

#define S_LEN  2048
#define NHEAD  12
#define HSZ    64
#define HDIM   768    // NH*HS == H
#define MROWS  8192   // B*S

__device__ __forceinline__ int lane_id() { return (int)(threadIdx.x & 31u); }

__device__ __forceinline__ v8f wmma_bf16(v16bf a, v16bf b, v8f c) {
  // D = A(16x32 bf16) * B(32x16 bf16) + C(16x16 f32)
  return __builtin_amdgcn_wmma_f32_16x16x32_bf16(
      /*neg_a=*/false, a, /*neg_b=*/false, b,
      /*c_mod=*/(short)0, c, /*reuse_a=*/false, /*reuse_b=*/false);
}

#if USE_ASYNC_LDS
// Builtin prototype (from hipcc diagnostic): both pointer params have pointee
// type 'int __attribute__((vector_size(16)))' in AS1 (global) / AS3 (LDS).
typedef int v4i_ __attribute__((vector_size(4 * sizeof(int))));
typedef __attribute__((address_space(1))) v4i_ as1_v4i;
typedef __attribute__((address_space(3))) v4i_ as3_v4i;

// Copy 16 contiguous bf16 (32B) global -> LDS with two async b128 transfers.
__device__ __forceinline__ void async_copy16(const __bf16* g, __bf16* l) {
  __builtin_amdgcn_global_load_async_to_lds_b128((as1_v4i*)g, (as3_v4i*)l, 0, 0);
  __builtin_amdgcn_global_load_async_to_lds_b128((as1_v4i*)g, (as3_v4i*)l, 16, 0);
}
#endif

__device__ __forceinline__ void copy16(const __bf16* __restrict__ g,
                                       __bf16* __restrict__ l) {
#if USE_ASYNC_LDS
  async_copy16(g, l);
#else
#pragma unroll
  for (int e = 0; e < 16; ++e) l[e] = g[e];
#endif
}

__device__ __forceinline__ void async_fence_stage() {
#if USE_ASYNC_LDS
  __builtin_amdgcn_s_wait_asynccnt(0);
#endif
  __syncthreads();
}

// A-fragment (16x32, row-major source, leading dim ld):
//   lane: m = lane&15, half = lane>>4 ; element e -> k = 16*half + e
__device__ __forceinline__ v16bf load_afrag(const __bf16* __restrict__ src, int ld) {
  int ln = lane_id();
  const __bf16* row = src + (size_t)(ln & 15) * ld;
  int kb = 16 * (ln >> 4);
  v16bf f;
#pragma unroll
  for (int e = 0; e < 16; ++e) f[e] = row[kb + e];
  return f;
}

// B-fragment (32x16) from an N-major (transposed) source: src[n*ld + k].
//   lane: n = lane&15, half = lane>>4 ; element e -> k = 16*half + e
//   -> one contiguous 16-element (32B) read per lane (2x ds_load_b128).
__device__ __forceinline__ v16bf load_bfrag_T(const __bf16* __restrict__ src, int ld) {
  int ln = lane_id();
  const __bf16* row = src + (size_t)(ln & 15) * ld + 16 * (ln >> 4);
  v16bf f;
#pragma unroll
  for (int e = 0; e < 16; ++e) f[e] = row[e];
  return f;
}

// ---------------------------------------------------------------------------
// fp32 -> bf16 conversion, 4 elements / thread
// ---------------------------------------------------------------------------
__global__ void cvt_f32_to_bf16(const float* __restrict__ in,
                                __bf16* __restrict__ out, int n) {
  int i = (blockIdx.x * blockDim.x + threadIdx.x) * 4;
  if (i + 3 < n) {
#pragma unroll
    for (int k = 0; k < 4; ++k) out[i + k] = (__bf16)in[i + k];
  } else {
    for (int k = 0; k < 4 && i + k < n; ++k) out[i + k] = (__bf16)in[i + k];
  }
}

// ---------------------------------------------------------------------------
// GEMM: C(MxN) = A(MxK bf16, row-major) * B(KxN bf16, row-major) + bias[n]
// Workgroup tile 128x128, 8 waves (4x2), each wave 32x64 (2x4 WMMA frags).
// A tile staged row-major (async path); B tile staged TRANSPOSED (n-major).
// HEADS=true: write bf16 into (B,NH,S,HS);  HEADS=false: write fp32 MxN.
// ---------------------------------------------------------------------------
template <bool HEADS>
__global__ __launch_bounds__(256) void gemm_bf16_kernel(
    const __bf16* __restrict__ A, const __bf16* __restrict__ Bm,
    const float* __restrict__ bias, void* __restrict__ Out,
    int M, int N, int K) {
  __shared__ __bf16 As[128 * 32];    // row-major  (m x k)
  __shared__ __bf16 BsT[128 * 32];   // n-major    (n x k)

  const int tid  = threadIdx.x;
  const int wave = tid >> 5;
  const int ln   = tid & 31;
  const int wr   = wave >> 1;      // 0..3  (row strip of 32)
  const int wc   = wave & 1;       // 0..1  (col strip of 64)
  const int tile_m = blockIdx.x * 128;
  const int tile_n = blockIdx.y * 128;

  const int ar = tid >> 1, ac = (tid & 1) * 16;   // A staging coords
  const int br = tid >> 3, bc = (tid & 7) * 16;   // B staging coords

  v8f acc[2][4];
#pragma unroll
  for (int i = 0; i < 2; ++i)
#pragma unroll
    for (int j = 0; j < 4; ++j)
#pragma unroll
      for (int r = 0; r < 8; ++r) acc[i][j][r] = 0.0f;

  for (int k0 = 0; k0 < K; k0 += 32) {
    __syncthreads();   // previous iteration's LDS reads complete

    if (k0 + 32 < K) { // prefetch next slices into cache
      __builtin_prefetch(A + (size_t)(tile_m + ar) * K + (k0 + 32) + ac, 0, 0);
      __builtin_prefetch(Bm + (size_t)(k0 + 32 + br) * N + tile_n + bc, 0, 0);
    }

    // stage A tile 128x32 row-major (16 contiguous bf16 per thread)
    copy16(A + (size_t)(tile_m + ar) * K + k0 + ac, &As[ar * 32 + ac]);

    // stage B tile 32x128 -> BsT (n-major): vector global load, strided LDS store
    {
      const __bf16* g = Bm + (size_t)(k0 + br) * N + tile_n + bc;
      __bf16 tmp[16];
#pragma unroll
      for (int e = 0; e < 16; ++e) tmp[e] = g[e];
#pragma unroll
      for (int e = 0; e < 16; ++e) BsT[(bc + e) * 32 + br] = tmp[e];
    }

    async_fence_stage();

    v16bf af0 = load_afrag(&As[(wr * 32 + 0) * 32], 32);
    v16bf af1 = load_afrag(&As[(wr * 32 + 16) * 32], 32);
#pragma unroll
    for (int j = 0; j < 4; ++j) {
      v16bf bf = load_bfrag_T(&BsT[(wc * 64 + j * 16) * 32], 32);
      acc[0][j] = wmma_bf16(af0, bf, acc[0][j]);
      acc[1][j] = wmma_bf16(af1, bf, acc[1][j]);
    }
  }

  // epilogue: C layout -> row m = r + 8*half, col n = lane&15
  const int l15  = ln & 15;
  const int half = ln >> 4;
#pragma unroll
  for (int i = 0; i < 2; ++i) {
#pragma unroll
    for (int j = 0; j < 4; ++j) {
      int n  = tile_n + wc * 64 + j * 16 + l15;
      float bv = bias[n];
#pragma unroll
      for (int r = 0; r < 8; ++r) {
        int m   = tile_m + wr * 32 + i * 16 + r + 8 * half;
        float v = acc[i][j][r] + bv;
        if (HEADS) {
          // (B,NH,S,HS) bf16 : b = m/S, s = m%S, h = n/HS, d = n%HS
          int b = m >> 11, s = m & (S_LEN - 1);
          int h = n >> 6,  d = n & (HSZ - 1);
          ((__bf16*)Out)[(((size_t)(b * NHEAD + h) << 11) + s) * HSZ + d] = (__bf16)v;
        } else {
          ((float*)Out)[(size_t)m * N + n] = v;
        }
      }
    }
  }
}

// ---------------------------------------------------------------------------
// Streaming (flash-style) attention.
// Grid: (S/64, B*NH). Block: 128 threads = 4 waves; wave w owns Q rows
// [blockIdx.x*64 + w*16, +16). K/V blocks (32 keys) staged cooperatively:
// K row-major (for S=Q*K^T), V transposed (feature-major, for O+=P*V).
// ---------------------------------------------------------------------------
__global__ __launch_bounds__(128) void attn_kernel(
    const __bf16* __restrict__ Q, const __bf16* __restrict__ K,
    const __bf16* __restrict__ V, const float* __restrict__ mask,
    __bf16* __restrict__ Cmb) {
  __shared__ __bf16 Ks[32 * 64];       // key-major   (k x d)
  __shared__ __bf16 VsT[64 * 32];      // feature-major (d x k)
  __shared__ __bf16 pls[4][16 * 32];   // per-wave P staging (C->A layout)

  const int tid  = threadIdx.x;
  const int wave = tid >> 5;
  const int ln   = tid & 31;
  const int l15  = ln & 15;
  const int half = ln >> 4;

  const int bh = blockIdx.y;
  const int b  = bh / NHEAD;
  const int h  = bh - b * NHEAD;
  const __bf16* Qh = Q + (size_t)bh * S_LEN * HSZ;
  const __bf16* Kh = K + (size_t)bh * S_LEN * HSZ;
  const __bf16* Vh = V + (size_t)bh * S_LEN * HSZ;
  const float*  mk = mask + (size_t)b * S_LEN;

  const int q0 = blockIdx.x * 64 + wave * 16;
  const int sr = tid >> 2, sc = (tid & 3) * 16;   // staging coords (32 x 64)

  // Q strip 16x64 as two A-fragments (HS split into two 32-deep chunks)
  v16bf qa0 = load_afrag(Qh + (size_t)q0 * HSZ + 0,  HSZ);
  v16bf qa1 = load_afrag(Qh + (size_t)q0 * HSZ + 32, HSZ);

  v8f o[4];
  float mrow[8], lrow[8];
#pragma unroll
  for (int j = 0; j < 4; ++j)
#pragma unroll
    for (int r = 0; r < 8; ++r) o[j][r] = 0.0f;
#pragma unroll
  for (int r = 0; r < 8; ++r) { mrow[r] = -3.0e38f; lrow[r] = 0.0f; }

  const float scale = 0.125f;  // 1/sqrt(64)

  for (int kb = 0; kb < S_LEN; kb += 32) {
    __syncthreads();   // previous block's LDS reads complete

    if (kb + 32 < S_LEN) {
      __builtin_prefetch(Kh + (size_t)(kb + 32 + sr) * HSZ + sc, 0, 0);
      __builtin_prefetch(Vh + (size_t)(kb + 32 + sr) * HSZ + sc, 0, 0);
    }

    // stage K block 32x64 row-major (async path)
    copy16(Kh + (size_t)(kb + sr) * HSZ + sc, &Ks[sr * 64 + sc]);

    // stage V block transposed: VsT[d*32 + k]
    {
      const __bf16* g = Vh + (size_t)(kb + sr) * HSZ + sc;
      __bf16 tmp[16];
#pragma unroll
      for (int e = 0; e < 16; ++e) tmp[e] = g[e];
#pragma unroll
      for (int e = 0; e < 16; ++e) VsT[(sc + e) * 32 + sr] = tmp[e];
    }

    async_fence_stage();

    // ---- scores S = Q * K^T for 32 keys: two 16x16 f32 fragments ----
    v8f s0, s1;
#pragma unroll
    for (int r = 0; r < 8; ++r) { s0[r] = 0.0f; s1[r] = 0.0f; }
    {
      v16bf b00 = load_bfrag_T(&Ks[(0)  * 64 + 0],  64);
      v16bf b01 = load_bfrag_T(&Ks[(0)  * 64 + 32], 64);
      v16bf b10 = load_bfrag_T(&Ks[(16) * 64 + 0],  64);
      v16bf b11 = load_bfrag_T(&Ks[(16) * 64 + 32], 64);
      s0 = wmma_bf16(qa0, b00, s0);
      s0 = wmma_bf16(qa1, b01, s0);
      s1 = wmma_bf16(qa0, b10, s1);
      s1 = wmma_bf16(qa1, b11, s1);
    }
    float mk0 = mk[kb + l15];
    float mk1 = mk[kb + 16 + l15];

    // ---- online softmax update (per lane: 8 rows, 2 columns) ----
#pragma unroll
    for (int r = 0; r < 8; ++r) {
      float a0 = s0[r] * scale + mk0;
      float a1 = s1[r] * scale + mk1;
      float mx = fmaxf(a0, a1);
#pragma unroll
      for (int off = 8; off >= 1; off >>= 1)
        mx = fmaxf(mx, __shfl_xor(mx, off, 32));   // stays within half (bits 0..3)
      float mnew = fmaxf(mrow[r], mx);
      float corr = __expf(mrow[r] - mnew);
      mrow[r] = mnew;
      float p0 = __expf(a0 - mnew);
      float p1 = __expf(a1 - mnew);
      float rs = p0 + p1;
#pragma unroll
      for (int off = 8; off >= 1; off >>= 1)
        rs += __shfl_xor(rs, off, 32);
      lrow[r] = lrow[r] * corr + rs;
#pragma unroll
      for (int j = 0; j < 4; ++j) o[j][r] *= corr;
      int row = r + 8 * half;
      pls[wave][row * 32 + l15]      = (__bf16)p0;
      pls[wave][row * 32 + 16 + l15] = (__bf16)p1;
    }

    // ---- O += P(16x32) * V(32x64): re-fragment P via LDS, 4 WMMAs ----
    v16bf pa = load_afrag(&pls[wave][0], 32);
#pragma unroll
    for (int j = 0; j < 4; ++j) {
      v16bf vf = load_bfrag_T(&VsT[(j * 16) * 32], 32);
      o[j] = wmma_bf16(pa, vf, o[j]);
    }
  }

  // ---- normalize and write combined (b, s, nh*hs) bf16 ----
#pragma unroll
  for (int r = 0; r < 8; ++r) {
    float inv = 1.0f / lrow[r];
    int srow = q0 + r + 8 * half;
#pragma unroll
    for (int j = 0; j < 4; ++j) {
      int col = h * HSZ + j * 16 + l15;
      Cmb[((size_t)b * S_LEN + srow) * HDIM + col] = (__bf16)(o[j][r] * inv);
    }
  }
}

// ---------------------------------------------------------------------------
// Host launch
// ---------------------------------------------------------------------------
extern "C" void kernel_launch(void* const* d_in, const int* in_sizes, int n_in,
                              void* d_out, int out_size, void* d_ws, size_t ws_size,
                              hipStream_t stream) {
  const float* x    = (const float*)d_in[0];
  const float* mask = (const float*)d_in[1];
  const float* Wq   = (const float*)d_in[2];
  const float* bq   = (const float*)d_in[3];
  const float* Wk   = (const float*)d_in[4];
  const float* bk   = (const float*)d_in[5];
  const float* Wv   = (const float*)d_in[6];
  const float* bv   = (const float*)d_in[7];
  const float* Wo   = (const float*)d_in[8];
  const float* bo   = (const float*)d_in[9];

  char* ws = (char*)d_ws;
  size_t off = 0;
  auto take = [&](size_t bytes) -> void* {
    void* p = ws + off;
    off += (bytes + 255) & ~(size_t)255;
    return p;
  };
  __bf16* xb  = (__bf16*)take((size_t)MROWS * HDIM * 2);
  __bf16* wqb = (__bf16*)take((size_t)HDIM * HDIM * 2);
  __bf16* wkb = (__bf16*)take((size_t)HDIM * HDIM * 2);
  __bf16* wvb = (__bf16*)take((size_t)HDIM * HDIM * 2);
  __bf16* wob = (__bf16*)take((size_t)HDIM * HDIM * 2);
  __bf16* Qb  = (__bf16*)take((size_t)MROWS * HDIM * 2);
  __bf16* Kb  = (__bf16*)take((size_t)MROWS * HDIM * 2);
  __bf16* Vb  = (__bf16*)take((size_t)MROWS * HDIM * 2);
  __bf16* cmb = (__bf16*)take((size_t)MROWS * HDIM * 2);

  auto cvt = [&](const float* src, __bf16* dst, int n) {
    int thr = (n + 3) / 4;
    cvt_f32_to_bf16<<<(thr + 255) / 256, 256, 0, stream>>>(src, dst, n);
  };
  cvt(x,  xb,  MROWS * HDIM);
  cvt(Wq, wqb, HDIM * HDIM);
  cvt(Wk, wkb, HDIM * HDIM);
  cvt(Wv, wvb, HDIM * HDIM);
  cvt(Wo, wob, HDIM * HDIM);

  dim3 gg(MROWS / 128, HDIM / 128);   // (64, 6)
  gemm_bf16_kernel<true><<<gg, 256, 0, stream>>>(xb, wqb, bq, Qb, MROWS, HDIM, HDIM);
  gemm_bf16_kernel<true><<<gg, 256, 0, stream>>>(xb, wkb, bk, Kb, MROWS, HDIM, HDIM);
  gemm_bf16_kernel<true><<<gg, 256, 0, stream>>>(xb, wvb, bv, Vb, MROWS, HDIM, HDIM);

  dim3 ga(S_LEN / 64, 4 * NHEAD);     // (32, 48)
  attn_kernel<<<ga, 128, 0, stream>>>(Qb, Kb, Vb, mask, cmb);

  gemm_bf16_kernel<false><<<gg, 256, 0, stream>>>(cmb, wob, bo, d_out, MROWS, HDIM, HDIM);

  (void)in_sizes; (void)n_in; (void)out_size; (void)ws_size;
}
